// Model_4544075399519
// MI455X (gfx1250) — matrix operand order repaired and checked by reading, not compiled, not run on hardware
//
#include <hip/hip_runtime.h>
#include <hip/hip_bf16.h>
#include <math.h>

// ---------------------------------------------------------------------------
// Model dims (fixed by the reference)
// ---------------------------------------------------------------------------
#define S_LEN   1024
#define HID     512
#define G4      2048            // 4*H
#define D_EMB   300
#define D_EMB_P 320             // padded to multiple of 32 for WMMA K
#define D2H     1024            // 2*H

typedef __bf16 bf16;
typedef __attribute__((ext_vector_type(16))) bf16  v16bf;
typedef __attribute__((ext_vector_type(8)))  bf16  v8bf;
typedef __attribute__((ext_vector_type(8)))  float v8f;

union AFrag { v16bf v; v8bf h[2]; };

// ---------------------------------------------------------------------------
// 1) Embedding gather + f32 -> bf16 with K padding (300 -> 320)
// ---------------------------------------------------------------------------
__global__ void gather_x_bf16(const int* __restrict__ X,
                              const float* __restrict__ embed,
                              bf16* __restrict__ out) {
    int idx = blockIdx.x * 256 + threadIdx.x;       // over 1024*320
    if (idx >= S_LEN * D_EMB_P) return;
    int i = idx / D_EMB_P, k = idx - i * D_EMB_P;
    float v = 0.0f;
    if (k < D_EMB) v = embed[(long)X[i] * D_EMB + k];
    out[idx] = (bf16)v;
}

// Generic f32 -> bf16 convert with optional K padding (Kin <= Kout)
__global__ void convert_pad_bf16(const float* __restrict__ src,
                                 bf16* __restrict__ dst,
                                 int rows, int Kin, int Kout) {
    long idx = (long)blockIdx.x * 256 + threadIdx.x;
    long total = (long)rows * Kout;
    if (idx >= total) return;
    int r = (int)(idx / Kout), k = (int)(idx - (long)r * Kout);
    float v = (k < Kin) ? src[(long)r * Kin + k] : 0.0f;
    dst[idx] = (bf16)v;
}

// ---------------------------------------------------------------------------
// 2) WMMA GEMM:  C[M,N] = A[M,K] * W[N,K]^T + b0[N] + b1[N]
//    A, W in bf16 (row-major), C f32. M%128==0, N%64==0, K%32==0.
//    Block = 256 threads (8 waves); block tile 128x64; wave tile 16x64.
// ---------------------------------------------------------------------------
__global__ void gemm_bf16_wmma_bias(const bf16* __restrict__ A,
                                    const bf16* __restrict__ W,
                                    const float* __restrict__ b0,
                                    const float* __restrict__ b1,
                                    float* __restrict__ C,
                                    int M, int N, int K) {
    const int tid   = threadIdx.x;
    const int wave  = tid >> 5;
    const int lane  = tid & 31;
    const int l16   = lane & 15;
    const int lhalf = lane >> 4;                    // 0: lanes 0-15, 1: 16-31

    const int rowBase = blockIdx.x * 128 + wave * 16;
    const int colBase = blockIdx.y * 64;

    v8f acc[4];
    #pragma unroll
    for (int t = 0; t < 4; ++t) acc[t] = (v8f)(0.0f);

    const int arow = rowBase + l16;
    const bf16* __restrict__ aRow = A + (size_t)arow * K;

    for (int kb = 0; kb < K; kb += 32) {
        // A fragment: lane<16 -> K [0..15]; lane>=16 -> K [16..31]
        AFrag af;
        const bf16* ap = aRow + kb + lhalf * 16;
        af.h[0] = *(const v8bf*)(ap);
        af.h[1] = *(const v8bf*)(ap + 8);

        #pragma unroll
        for (int nt = 0; nt < 4; ++nt) {
            const int col = colBase + nt * 16 + l16;
            // B fragment: contiguous 16 bf16 of W row `col`, K offset by half-wave
            const bf16* wp = W + (size_t)col * K + kb + lhalf * 16;
            v16bf bfrag = *(const v16bf*)(wp);
            acc[nt] = __builtin_amdgcn_wmma_f32_16x16x32_bf16(
                false, af.v, false, bfrag, (short)0, acc[nt], false, false);
        }
    }

    // Epilogue: documented C layout — VGPR r: lanes 0-15 M=r, lanes 16-31 M=8+r
    #pragma unroll
    for (int nt = 0; nt < 4; ++nt) {
        const int col = colBase + nt * 16 + l16;
        const float bias = b0[col] + b1[col];
        #pragma unroll
        for (int r = 0; r < 8; ++r) {
            const int row = rowBase + r + lhalf * 8;
            C[(size_t)row * N + col] = acc[nt][r] + bias;
        }
    }
}

// ---------------------------------------------------------------------------
// 3) Persistent bidirectional LSTM recurrence (one workgroup per direction).
//    blockDim = 1024: thread t (j = t&511):
//      half 0 computes gate rows j (i) and 1024+j (g) and owns c[j]/h[j]
//      half 1 computes gate rows 512+j (f) and 1536+j (o), posts via LDS
//    whh streamed from L2 each step; h state lives in LDS.
// ---------------------------------------------------------------------------
__device__ __forceinline__ float sigmoidf_(float x) {
    return 1.0f / (1.0f + __expf(-x));
}

__global__ void lstm_rec(const float* __restrict__ xpF,
                         const float* __restrict__ xpB,
                         const float* __restrict__ whhF,
                         const float* __restrict__ whhB,
                         float* __restrict__ hcat) {
    const int dir = blockIdx.x;                     // 0 = forward, 1 = backward
    const float* __restrict__ xp  = dir ? xpB  : xpF;
    const float* __restrict__ whh = dir ? whhB : whhF;
    const int colOff = dir * HID;

    __shared__ __align__(16) float hs[HID];
    __shared__ float gf[HID];
    __shared__ float go[HID];

    const int tid = threadIdx.x;
    const int j   = tid & (HID - 1);
    const int hi  = tid >> 9;                       // 0 or 1
    const int r0  = hi ? (HID + j)     : j;         // f : i
    const int r1  = hi ? (3 * HID + j) : (2 * HID + j); // o : g

    if (!hi) hs[j] = 0.0f;
    float c = 0.0f;
    __syncthreads();

    const float4* __restrict__ wr0 = (const float4*)(whh + (size_t)r0 * HID);
    const float4* __restrict__ wr1 = (const float4*)(whh + (size_t)r1 * HID);
    const float4* __restrict__ hv  = (const float4*)hs;

    for (int s = 0; s < S_LEN; ++s) {
        const int t = dir ? (S_LEN - 1 - s) : s;
        float a0 = xp[(size_t)t * G4 + r0];
        float a1 = xp[(size_t)t * G4 + r1];

        if (s + 1 < S_LEN) {                        // prefetch next step's xproj row
            const int tn = dir ? (t - 1) : (t + 1);
            __builtin_prefetch(&xp[(size_t)tn * G4 + r0], 0, 3);
        }

        #pragma unroll 4
        for (int k = 0; k < HID / 4; ++k) {
            const float4 h4 = hv[k];
            const float4 w0 = wr0[k];
            const float4 w1 = wr1[k];
            a0 = fmaf(h4.x, w0.x, a0); a0 = fmaf(h4.y, w0.y, a0);
            a0 = fmaf(h4.z, w0.z, a0); a0 = fmaf(h4.w, w0.w, a0);
            a1 = fmaf(h4.x, w1.x, a1); a1 = fmaf(h4.y, w1.y, a1);
            a1 = fmaf(h4.z, w1.z, a1); a1 = fmaf(h4.w, w1.w, a1);
        }

        if (hi) { gf[j] = a0; go[j] = a1; }
        __syncthreads();

        if (!hi) {
            const float ig = sigmoidf_(a0);
            const float gg = tanhf(a1);
            const float fg = sigmoidf_(gf[j]);
            const float og = sigmoidf_(go[j]);
            c = fg * c + ig * gg;
            const float hn = og * tanhf(c);
            hs[j] = hn;
            hcat[(size_t)t * D2H + colOff + j] = hn;
        }
        __syncthreads();
    }
}

// ---------------------------------------------------------------------------
// 4) left/right projections: left[i,k]=h[i,:]·w1[k,:1024], right uses w1[k,1024:]
//    One 32-thread wave per row i: lanes 0-15 -> left, 16-31 -> right.
// ---------------------------------------------------------------------------
__global__ void lr_proj(const float* __restrict__ h,
                        const float* __restrict__ w1,
                        float* __restrict__ left,
                        float* __restrict__ right) {
    const int i   = blockIdx.x;
    const int tid = threadIdx.x;                    // 0..31
    const int k   = tid & 15;
    const float4* __restrict__ hr = (const float4*)(h + (size_t)i * D2H);
    const float4* __restrict__ wr =
        (const float4*)(w1 + (size_t)k * G4 + ((tid < 16) ? 0 : D2H));
    float s = 0.0f;
    #pragma unroll 4
    for (int q = 0; q < D2H / 4; ++q) {
        const float4 a = hr[q], b = wr[q];
        s = fmaf(a.x, b.x, s); s = fmaf(a.y, b.y, s);
        s = fmaf(a.z, b.z, s); s = fmaf(a.w, b.w, s);
    }
    if (tid < 16) left[i * 16 + k] = s;
    else          right[i * 16 + k] = s;
}

// ---------------------------------------------------------------------------
// 5) Fused score + row softmax:
//    out[i,j] = softmax_j( sum_k relu(left[i,k]+right[j,k]+b1[k])*w2[k] + b2 )
// ---------------------------------------------------------------------------
__global__ void attn_softmax(const float* __restrict__ left,
                             const float* __restrict__ right,
                             const float* __restrict__ b1,
                             const float* __restrict__ w2,
                             const float* __restrict__ b2,
                             float* __restrict__ out) {
    __shared__ float lb[16];
    __shared__ float w2s[16];
    __shared__ float sc[S_LEN];
    __shared__ float red[256];

    const int i   = blockIdx.x;
    const int tid = threadIdx.x;                    // 256 threads
    if (tid < 16) { lb[tid] = left[i * 16 + tid] + b1[tid]; w2s[tid] = w2[tid]; }
    __syncthreads();

    const float bb = b2[0];
    for (int j = tid; j < S_LEN; j += 256) {
        float s = bb;
        const float* rr = right + (size_t)j * 16;
        #pragma unroll
        for (int k = 0; k < 16; ++k) {
            const float v = lb[k] + rr[k];
            s = fmaf(w2s[k], (v > 0.0f ? v : 0.0f), s);
        }
        sc[j] = s;
    }
    __syncthreads();

    float m = -INFINITY;
    for (int j = tid; j < S_LEN; j += 256) m = fmaxf(m, sc[j]);
    red[tid] = m; __syncthreads();
    for (int o = 128; o > 0; o >>= 1) {
        if (tid < o) red[tid] = fmaxf(red[tid], red[tid + o]);
        __syncthreads();
    }
    m = red[0]; __syncthreads();

    float lsum = 0.0f;
    for (int j = tid; j < S_LEN; j += 256) {
        const float e = __expf(sc[j] - m);
        sc[j] = e; lsum += e;
    }
    red[tid] = lsum; __syncthreads();
    for (int o = 128; o > 0; o >>= 1) {
        if (tid < o) red[tid] += red[tid + o];
        __syncthreads();
    }
    const float inv = 1.0f / red[0];
    for (int j = tid; j < S_LEN; j += 256)
        out[(size_t)i * S_LEN + j] = sc[j] * inv;
}

// ---------------------------------------------------------------------------
// Host launcher
// ---------------------------------------------------------------------------
static inline size_t alignUp(size_t x) { return (x + 255) & ~(size_t)255; }

extern "C" void kernel_launch(void* const* d_in, const int* in_sizes, int n_in,
                              void* d_out, int out_size, void* d_ws, size_t ws_size,
                              hipStream_t stream) {
    (void)in_sizes; (void)n_in; (void)out_size; (void)ws_size;

    const int*   X     = (const int*)  d_in[0];
    // d_in[1] = P (unused by the reference)
    const float* embed = (const float*)d_in[2];
    const float* wih0f = (const float*)d_in[3];
    const float* whh0f = (const float*)d_in[4];
    const float* bih0f = (const float*)d_in[5];
    const float* bhh0f = (const float*)d_in[6];
    const float* wih0b = (const float*)d_in[7];
    const float* whh0b = (const float*)d_in[8];
    const float* bih0b = (const float*)d_in[9];
    const float* bhh0b = (const float*)d_in[10];
    const float* wih1f = (const float*)d_in[11];
    const float* whh1f = (const float*)d_in[12];
    const float* bih1f = (const float*)d_in[13];
    const float* bhh1f = (const float*)d_in[14];
    const float* wih1b = (const float*)d_in[15];
    const float* whh1b = (const float*)d_in[16];
    const float* bih1b = (const float*)d_in[17];
    const float* bhh1b = (const float*)d_in[18];
    const float* w1    = (const float*)d_in[19];
    const float* b1    = (const float*)d_in[20];
    const float* w2    = (const float*)d_in[21];
    const float* b2    = (const float*)d_in[22];
    float* out = (float*)d_out;

    // Workspace carve-up
    char* ws = (char*)d_ws;
    size_t off = 0;
    auto carve = [&](size_t bytes) { char* p = ws + off; off += alignUp(bytes); return p; };

    bf16*  xbf   = (bf16*) carve((size_t)S_LEN * D_EMB_P * sizeof(bf16));
    bf16*  w0fB  = (bf16*) carve((size_t)G4 * D_EMB_P * sizeof(bf16));
    bf16*  w0bB  = (bf16*) carve((size_t)G4 * D_EMB_P * sizeof(bf16));
    bf16*  w1fB  = (bf16*) carve((size_t)G4 * D2H * sizeof(bf16));
    bf16*  w1bB  = (bf16*) carve((size_t)G4 * D2H * sizeof(bf16));
    float* xp0f  = (float*)carve((size_t)S_LEN * G4 * sizeof(float));
    float* xp0b  = (float*)carve((size_t)S_LEN * G4 * sizeof(float));
    float* hcat0 = (float*)carve((size_t)S_LEN * D2H * sizeof(float));
    bf16*  h0bf  = (bf16*) carve((size_t)S_LEN * D2H * sizeof(bf16));
    float* xp1f  = (float*)carve((size_t)S_LEN * G4 * sizeof(float));
    float* xp1b  = (float*)carve((size_t)S_LEN * G4 * sizeof(float));
    float* hcat1 = (float*)carve((size_t)S_LEN * D2H * sizeof(float));
    float* leftB = (float*)carve((size_t)S_LEN * 16 * sizeof(float));
    float* rightB= (float*)carve((size_t)S_LEN * 16 * sizeof(float));

    // 1) gather + convert inputs to bf16
    {
        int n = S_LEN * D_EMB_P;
        gather_x_bf16<<<(n + 255) / 256, 256, 0, stream>>>(X, embed, xbf);
    }
    {
        long n = (long)G4 * D_EMB_P;
        dim3 g((unsigned)((n + 255) / 256));
        convert_pad_bf16<<<g, 256, 0, stream>>>(wih0f, w0fB, G4, D_EMB, D_EMB_P);
        convert_pad_bf16<<<g, 256, 0, stream>>>(wih0b, w0bB, G4, D_EMB, D_EMB_P);
    }
    {
        long n = (long)G4 * D2H;
        dim3 g((unsigned)((n + 255) / 256));
        convert_pad_bf16<<<g, 256, 0, stream>>>(wih1f, w1fB, G4, D2H, D2H);
        convert_pad_bf16<<<g, 256, 0, stream>>>(wih1b, w1bB, G4, D2H, D2H);
    }

    // 2) layer-0 input projections (WMMA)
    {
        dim3 g(S_LEN / 128, G4 / 64);
        gemm_bf16_wmma_bias<<<g, 256, 0, stream>>>(xbf, w0fB, bih0f, bhh0f, xp0f,
                                                   S_LEN, G4, D_EMB_P);
        gemm_bf16_wmma_bias<<<g, 256, 0, stream>>>(xbf, w0bB, bih0b, bhh0b, xp0b,
                                                   S_LEN, G4, D_EMB_P);
    }

    // 3) layer-0 recurrence (persistent, 2 workgroups = 2 directions)
    lstm_rec<<<2, 1024, 0, stream>>>(xp0f, xp0b, whh0f, whh0b, hcat0);

    // 4) convert layer-0 output to bf16, layer-1 projections (WMMA)
    {
        long n = (long)S_LEN * D2H;
        dim3 g((unsigned)((n + 255) / 256));
        convert_pad_bf16<<<g, 256, 0, stream>>>(hcat0, h0bf, S_LEN, D2H, D2H);
    }
    {
        dim3 g(S_LEN / 128, G4 / 64);
        gemm_bf16_wmma_bias<<<g, 256, 0, stream>>>(h0bf, w1fB, bih1f, bhh1f, xp1f,
                                                   S_LEN, G4, D2H);
        gemm_bf16_wmma_bias<<<g, 256, 0, stream>>>(h0bf, w1bB, bih1b, bhh1b, xp1b,
                                                   S_LEN, G4, D2H);
    }

    // 5) layer-1 recurrence
    lstm_rec<<<2, 1024, 0, stream>>>(xp1f, xp1b, whh1f, whh1b, hcat1);

    // 6) left/right projections
    lr_proj<<<S_LEN, 32, 0, stream>>>(hcat1, w1, leftB, rightB);

    // 7) fused score + softmax -> d_out
    attn_softmax<<<S_LEN, 256, 0, stream>>>(leftB, rightB, b1, w2, b2, out);
}
